// AttentionDecoder_90666759618636
// MI455X (gfx1250) — matrix operand order repaired and checked by reading, not compiled, not run on hardware
//
#include <hip/hip_runtime.h>
#include <hip/hip_bf16.h>

// ---------------------------------------------------------------------------
// MI455X (gfx1250) attention-LSTM decoder.
// bf16 WMMA (v_wmma_f32_16x16x32_bf16) for all GEMMs, fp32 accumulation;
// fp32 elementwise with hardware v_tanh_f32 where available.
// ---------------------------------------------------------------------------

typedef __attribute__((ext_vector_type(16))) __bf16 v16bf;
typedef __attribute__((ext_vector_type(8)))  float  v8f;

#define Bsz    16
#define TENC   256
#define TDEC   64
#define ENCD   512
#define EMBD   256
#define HIDD   512
#define VOCAB  32000
#define PHON   41
#define NSTEP  63          // T_DEC - 1
#define MROWS  (NSTEP*Bsz) // 1008 rows of h2
#define MPAD   1024        // padded to 64 M-tiles for 2x2 register tiling

#if __has_builtin(__builtin_amdgcn_tanhf)
#define fast_tanh(x) __builtin_amdgcn_tanhf(x)
#else
#define fast_tanh(x) tanhf(x)
#endif

// ----------------------------- WMMA fragment helpers -----------------------
// A fragment (16x32 bf16): lane L holds row M=L%16; contiguous K range
// [ (L/16)*16, +16 ) (matches B fragment order).
__device__ __forceinline__ v16bf load_a_frag(const __bf16* A, int lda,
                                             int m0, int k0, int lane) {
  union { v16bf v; unsigned int u[8]; } f;
  const unsigned int* row = (const unsigned int*)
      (A + (m0 + (lane & 15)) * lda + k0 + ((lane >> 4) << 4));
#pragma unroll
  for (int r = 0; r < 8; ++r) {
    int kp = r << 1;   // 0,2,...,14
    f.u[r] = row[kp >> 1];
  }
  return f.v;
}

// B fragment (32x16 bf16): lane L holds column N=L%16 (a ROW of W for
// out = X @ W^T), contiguous K range [ (L/16)*16, +16 ) -> one 32B load.
__device__ __forceinline__ v16bf load_b_frag(const __bf16* W, int ldw,
                                             int n0, int k0, int lane) {
  return *(const v16bf*)(W + (n0 + (lane & 15)) * ldw + k0 + ((lane >> 4) << 4));
}

#define WMMA_BF16(a, b, c) \
  __builtin_amdgcn_wmma_f32_16x16x32_bf16(false, (a), false, (b), (short)0, \
                                          (c), false, false)

// D = A(16xK) * W(16 rows x K)^T + D, K multiple of 32.
__device__ __forceinline__ void gemm_frag(const __bf16* A, int lda, int m0,
                                          const __bf16* W, int ldw, int n0,
                                          int K, int lane, v8f& acc) {
  for (int k0 = 0; k0 < K; k0 += 32) {
    v16bf a = load_a_frag(A, lda, m0, k0, lane);
    v16bf b = load_b_frag(W, ldw, n0, k0, lane);
    acc = WMMA_BF16(a, b, acc);
  }
}

__device__ __forceinline__ float sigmoidf_(float x) {
  return 1.0f / (1.0f + __expf(-x));
}

// ----------------------------- small utility kernels -----------------------
__global__ void cvt_bf16_kernel(const float* __restrict__ src,
                                __bf16* __restrict__ dst, int n) {
  int i = blockIdx.x * 256 + threadIdx.x;
  int stride = gridDim.x * 256;
  for (; i < n; i += stride) dst[i] = (__bf16)src[i];
}

__global__ void init_kernel(__bf16* h_bf, float* c, __bf16* h2pad,
                            unsigned int* bar) {
  int i = blockIdx.x * 256 + threadIdx.x;           // 8192 threads
  if (i < Bsz * HIDD) { h_bf[i] = (__bf16)0.0f; c[i] = 0.0f; }
  h2pad[i] = (__bf16)0.0f;                          // zero rows 1008..1023
  if (i < 2) bar[i] = 0u;
}

// ----------------------------- enc_proj GEMM -------------------------------
// enc_proj(4096x512) = enc(4096x512) @ W_enc^T + b_enc ; wave per 16x16 tile.
__global__ void encproj_kernel(const __bf16* __restrict__ enc_bf,
                               const __bf16* __restrict__ Wenc_bf,
                               const float* __restrict__ b_enc,
                               float* __restrict__ enc_proj) {
  int gw   = (blockIdx.x * blockDim.x + threadIdx.x) >> 5;  // 8192 waves
  int lane = threadIdx.x & 31;
  int mt = gw >> 5;          // 256 M tiles
  int nt = gw & 31;          // 32  N tiles
  int m0 = mt * 16, n0 = nt * 16;
  v8f acc = {};
  gemm_frag(enc_bf, ENCD, m0, Wenc_bf, ENCD, n0, ENCD, lane, acc);
  int col = n0 + (lane & 15);
  float bias = b_enc[col];
  int rb = (lane >> 4) << 3;
#pragma unroll
  for (int r = 0; r < 8; ++r)
    enc_proj[(m0 + rb + r) * HIDD + col] = acc[r] + bias;
}

// ----------------------------- cooperative grid barrier --------------------
__device__ __forceinline__ void grid_barrier(unsigned int* bar, unsigned nblk) {
  __syncthreads();
  if (threadIdx.x == 0) {
    unsigned gen = __hip_atomic_load(&bar[1], __ATOMIC_RELAXED,
                                     __HIP_MEMORY_SCOPE_AGENT);
    unsigned prev = __hip_atomic_fetch_add(&bar[0], 1u, __ATOMIC_ACQ_REL,
                                           __HIP_MEMORY_SCOPE_AGENT);
    if (prev == nblk - 1u) {
      __hip_atomic_store(&bar[0], 0u, __ATOMIC_RELAXED,
                         __HIP_MEMORY_SCOPE_AGENT);
      __hip_atomic_fetch_add(&bar[1], 1u, __ATOMIC_RELEASE,
                             __HIP_MEMORY_SCOPE_AGENT);
    } else {
      while (__hip_atomic_load(&bar[1], __ATOMIC_ACQUIRE,
                               __HIP_MEMORY_SCOPE_AGENT) == gen)
        __builtin_amdgcn_s_sleep(1);
    }
  }
  __syncthreads();
}

// ----------------------------- persistent recurrence kernel ----------------
// 16 blocks x 256 threads (128 waves total). Per step:
//  phase1: q = h @ W_q^T + b_q            (waves 0..31, WMMA)
//  phase2: attention + softmax + ctx + emb (block b handles batch b) -> x_bf
//  phase3: gates = [x|h] @ [W_ih|W_hh]^T  (all 128 waves = 128 N tiles, WMMA)
//  phase4: LSTM cell elementwise -> h_bf, c, h2all_bf
__global__ void __launch_bounds__(256)
decoder_kernel(const float* __restrict__ enc,            // (16,256,512)
               const unsigned char* __restrict__ mask,   // (16,256)
               const int* __restrict__ targets,          // (16,64)
               const float* __restrict__ emb_table,      // (32000,256)
               const float* __restrict__ b_ih,
               const float* __restrict__ b_hh,
               const float* __restrict__ b_q,
               const float* __restrict__ v_vec,
               const __bf16* __restrict__ Wq_bf,
               const __bf16* __restrict__ Wih_bf,
               const __bf16* __restrict__ Whh_bf,
               const float* __restrict__ enc_proj,
               float* __restrict__ q_g,                  // (16,512) f32
               __bf16* __restrict__ x_bf,                // (16,768)
               __bf16* __restrict__ h_bf,                // (16,512)
               float* __restrict__ c_g,                  // (16,512)
               float* __restrict__ gates,                // (16,2048)
               __bf16* __restrict__ h2all,               // (1024,512) padded
               unsigned int* __restrict__ bar) {
  const int b     = blockIdx.x;          // batch handled in attention phase
  const int tid   = threadIdx.x;
  const int lane  = tid & 31;
  const int wave  = tid >> 5;
  const int gwave = b * 8 + wave;        // 0..127 global wave id

  __shared__ float q_s[HIDD];
  __shared__ float v_s[HIDD];
  __shared__ float attn_s[TENC];
  __shared__ float red[256];

  v_s[tid]       = v_vec[tid];
  v_s[tid + 256] = v_vec[tid + 256];
  __syncthreads();

  for (int t = 0; t < NSTEP; ++t) {
    // ---- phase 1: q GEMM (M=16,N=512,K=512) over waves 0..31 -------------
    if (gwave < 32) {
      int n0 = gwave * 16;
      v8f acc = {};
      gemm_frag(h_bf, HIDD, 0, Wq_bf, HIDD, n0, HIDD, lane, acc);
      int col = n0 + (lane & 15);
      float bias = b_q[col];
      int rb = (lane >> 4) << 3;
#pragma unroll
      for (int r = 0; r < 8; ++r) q_g[(rb + r) * HIDD + col] = acc[r] + bias;
    }
    grid_barrier(bar, Bsz);

    // ---- phase 2: attention for batch b ----------------------------------
    q_s[tid]       = q_g[b * HIDD + tid];
    q_s[tid + 256] = q_g[b * HIDD + 256 + tid];
    __syncthreads();

    // score[b, tid] = sum_j v[j] * tanh(enc_proj[b,tid,j] + q[b,j])
    float s = 0.0f;
    const float* ep = enc_proj + (b * TENC + tid) * HIDD;
#pragma unroll 4
    for (int j = 0; j < HIDD; ++j) s += v_s[j] * fast_tanh(ep[j] + q_s[j]);
    if (mask[b * TENC + tid]) s = -1e9f;

    // softmax over 256 positions
    red[tid] = s; __syncthreads();
    for (int off = 128; off; off >>= 1) {
      if (tid < off) red[tid] = fmaxf(red[tid], red[tid + off]);
      __syncthreads();
    }
    float mx = red[0]; __syncthreads();
    float e = __expf(s - mx);
    red[tid] = e; __syncthreads();
    for (int off = 128; off; off >>= 1) {
      if (tid < off) red[tid] += red[tid + off];
      __syncthreads();
    }
    float inv = 1.0f / red[0]; __syncthreads();
    attn_s[tid] = e * inv;
    __syncthreads();

    // ctx[b, e] = sum_t attn[t] * enc[b,t,e] ; coalesced over e
    float a0 = 0.0f, a1 = 0.0f;
    const float* eb = enc + (b * TENC) * ENCD;
    for (int tt = 0; tt < TENC; ++tt) {
      float w = attn_s[tt];
      a0 += w * eb[tt * ENCD + tid];
      a1 += w * eb[tt * ENCD + 256 + tid];
    }

    // embedding (padding_idx = 0), clipped token
    int tok = targets[b * TDEC + t];
    tok = tok < 0 ? 0 : (tok > VOCAB - 1 ? VOCAB - 1 : tok);
    float em = (tok != 0) ? emb_table[tok * EMBD + tid] : 0.0f;

    // x = [emb(256) | ctx(512)] as bf16
    x_bf[b * 768 + tid]       = (__bf16)em;
    x_bf[b * 768 + 256 + tid] = (__bf16)a0;
    x_bf[b * 768 + 512 + tid] = (__bf16)a1;
    grid_barrier(bar, Bsz);

    // ---- phase 3: gates GEMM (M=16,N=2048,K=768+512); wave -> N tile -----
    {
      int n0 = gwave * 16;
      v8f acc = {};
      gemm_frag(x_bf, 768, 0, Wih_bf, 768, n0, 768, lane, acc);
      gemm_frag(h_bf, HIDD, 0, Whh_bf, HIDD, n0, HIDD, lane, acc);
      int col = n0 + (lane & 15);
      float bias = b_ih[col] + b_hh[col];
      int rb = (lane >> 4) << 3;
#pragma unroll
      for (int r = 0; r < 8; ++r) gates[(rb + r) * 2048 + col] = acc[r] + bias;
    }
    grid_barrier(bar, Bsz);

    // ---- phase 4: LSTM cell elementwise (8192 elems / 4096 threads) ------
    int gtid = b * 256 + tid;
#pragma unroll
    for (int idx = gtid; idx < Bsz * HIDD; idx += Bsz * 256) {
      int bb = idx >> 9, j = idx & 511;
      const float* g = gates + bb * 2048;
      float ig = sigmoidf_(g[j]);
      float fg = sigmoidf_(g[512 + j]);
      float gg = fast_tanh(g[1024 + j]);
      float og = sigmoidf_(g[1536 + j]);
      float cp = c_g[idx];
      float c2 = fg * cp + ig * gg;
      float h2 = og * fast_tanh(c2);
      c_g[idx] = c2;
      h_bf[idx] = (__bf16)h2;
      h2all[(t * Bsz + bb) * HIDD + j] = (__bf16)h2;
    }
    grid_barrier(bar, Bsz);
  }
}

// ----------------------------- output projections --------------------------
// logits: (1008 x 32000) = h2all @ W_out^T + b_out, 2x2 register tiling:
// each wave computes a 32x32 patch (4 accumulators), so each k-step does
// 4 WMMAs on 2 A-frags + 2 B-frags -> 2 b128 loads per WMMA instead of 4.
// h2all is zero-padded to 1024 rows; second M-tile store is guarded.
__global__ void logits_kernel(const __bf16* __restrict__ h2,
                              const __bf16* __restrict__ Wout,
                              const float* __restrict__ b_out,
                              float* __restrict__ out) {
  int wave = threadIdx.x >> 5, lane = threadIdx.x & 31;
  int np = blockIdx.x % (VOCAB / 32);      // 1000 N-pairs
  int mg = blockIdx.x / (VOCAB / 32);      // 4 groups of 8 M-pairs
  int mp = mg * 8 + wave;                  // 0..31 M-pair
  int m0 = mp * 32;
  int n0 = np * 32;
  v8f acc00 = {}, acc01 = {}, acc10 = {}, acc11 = {};
  for (int k0 = 0; k0 < HIDD; k0 += 32) {
    if (k0 + 64 < HIDD) {                  // prefetch W_out 2 k-steps ahead
      __builtin_prefetch((const void*)(Wout + (n0 + (lane & 15)) * HIDD +
                                       k0 + 64), 0, 3);
      __builtin_prefetch((const void*)(Wout + (n0 + 16 + (lane & 15)) * HIDD +
                                       k0 + 64), 0, 3);
    }
    v16bf a0 = load_a_frag(h2, HIDD, m0,      k0, lane);
    v16bf a1 = load_a_frag(h2, HIDD, m0 + 16, k0, lane);
    v16bf b0 = load_b_frag(Wout, HIDD, n0,      k0, lane);
    v16bf b1 = load_b_frag(Wout, HIDD, n0 + 16, k0, lane);
    acc00 = WMMA_BF16(a0, b0, acc00);
    acc01 = WMMA_BF16(a0, b1, acc01);
    acc10 = WMMA_BF16(a1, b0, acc10);
    acc11 = WMMA_BF16(a1, b1, acc11);
  }
  int nlo = lane & 15;
  int rb  = (lane >> 4) << 3;
  float bias0 = b_out[n0 + nlo];
  float bias1 = b_out[n0 + 16 + nlo];
  int mt0 = mp * 2, mt1 = mp * 2 + 1;      // time indices (M tiles)
#pragma unroll
  for (int r = 0; r < 8; ++r) {
    int bb = rb + r;                       // batch index within tile
    float* o0 = out + (size_t)(bb * NSTEP + mt0) * VOCAB;
    o0[n0 + nlo]      = acc00[r] + bias0;
    o0[n0 + 16 + nlo] = acc01[r] + bias1;
  }
  if (mt1 < NSTEP) {
#pragma unroll
    for (int r = 0; r < 8; ++r) {
      int bb = rb + r;
      float* o1 = out + (size_t)(bb * NSTEP + mt1) * VOCAB;
      o1[n0 + nlo]      = acc10[r] + bias0;
      o1[n0 + 16 + nlo] = acc11[r] + bias1;
    }
  }
}

__global__ void phon_kernel(const __bf16* __restrict__ h2,
                            const __bf16* __restrict__ Wph,   // padded 48 rows
                            const float* __restrict__ b_ph,
                            float* __restrict__ out_ph) {
  int gw = (blockIdx.x * blockDim.x + threadIdx.x) >> 5;
  int lane = threadIdx.x & 31;
  if (gw >= NSTEP * 3) return;             // wave-uniform predicate
  int mt = gw / 3, nt = gw % 3;
  int n0 = nt * 16;
  v8f acc = {};
  gemm_frag(h2, HIDD, mt * 16, Wph, HIDD, n0, HIDD, lane, acc);
  int n = n0 + (lane & 15);
  if (n < PHON) {
    float bias = b_ph[n];
    int rb = (lane >> 4) << 3;
#pragma unroll
    for (int r = 0; r < 8; ++r) {
      int bb = rb + r;
      out_ph[(bb * NSTEP + mt) * PHON + n] = acc[r] + bias;
    }
  }
}

// ----------------------------- host launcher -------------------------------
extern "C" void kernel_launch(void* const* d_in, const int* in_sizes, int n_in,
                              void* d_out, int out_size, void* d_ws,
                              size_t ws_size, hipStream_t stream) {
  const float*         enc       = (const float*)d_in[0];
  const unsigned char* mask      = (const unsigned char*)d_in[1];
  const int*           targets   = (const int*)d_in[2];
  const float*         emb_table = (const float*)d_in[3];
  const float*         W_ih      = (const float*)d_in[4];
  const float*         b_ih      = (const float*)d_in[5];
  const float*         W_hh      = (const float*)d_in[6];
  const float*         b_hh      = (const float*)d_in[7];
  const float*         W_enc     = (const float*)d_in[8];
  const float*         b_enc     = (const float*)d_in[9];
  const float*         W_q       = (const float*)d_in[10];
  const float*         b_q       = (const float*)d_in[11];
  const float*         v_vec     = (const float*)d_in[12];
  const float*         W_out     = (const float*)d_in[13];
  const float*         b_out     = (const float*)d_in[14];
  const float*         W_ph      = (const float*)d_in[15];
  const float*         b_ph      = (const float*)d_in[16];

  float* out_logits = (float*)d_out;                      // (16,63,32000)
  float* out_phon   = out_logits + (size_t)Bsz * NSTEP * VOCAB;

  // workspace layout (256B aligned segments)
  size_t off = 0;
  auto take = [&](size_t bytes) {
    size_t o = off; off += (bytes + 255) & ~(size_t)255; return o;
  };
  char* ws = (char*)d_ws;
  __bf16* enc_bf  = (__bf16*)(ws + take((size_t)Bsz*TENC*ENCD*2));
  __bf16* Wq_bf   = (__bf16*)(ws + take((size_t)HIDD*HIDD*2));
  __bf16* Wih_bf  = (__bf16*)(ws + take((size_t)2048*768*2));
  __bf16* Whh_bf  = (__bf16*)(ws + take((size_t)2048*HIDD*2));
  __bf16* Wenc_bf = (__bf16*)(ws + take((size_t)HIDD*ENCD*2));
  __bf16* Wout_bf = (__bf16*)(ws + take((size_t)VOCAB*HIDD*2));
  __bf16* Wph_bf  = (__bf16*)(ws + take((size_t)48*HIDD*2));   // padded rows
  float*  eproj   = (float*) (ws + take((size_t)Bsz*TENC*HIDD*4));
  float*  q_g     = (float*) (ws + take((size_t)Bsz*HIDD*4));
  __bf16* x_bf    = (__bf16*)(ws + take((size_t)Bsz*768*2));
  __bf16* h_bf    = (__bf16*)(ws + take((size_t)Bsz*HIDD*2));
  float*  c_g     = (float*) (ws + take((size_t)Bsz*HIDD*4));
  float*  gates   = (float*) (ws + take((size_t)Bsz*2048*4));
  __bf16* h2all   = (__bf16*)(ws + take((size_t)MPAD*HIDD*2)); // padded 1024
  unsigned int* bar = (unsigned int*)(ws + take(256));

  auto cvt = [&](const float* s, __bf16* d, int n) {
    int g = (n + 255) / 256; if (g > 8192) g = 8192;
    cvt_bf16_kernel<<<g, 256, 0, stream>>>(s, d, n);
  };
  cvt(enc,   enc_bf,  Bsz*TENC*ENCD);
  cvt(W_q,   Wq_bf,   HIDD*HIDD);
  cvt(W_ih,  Wih_bf,  2048*768);
  cvt(W_hh,  Whh_bf,  2048*HIDD);
  cvt(W_enc, Wenc_bf, HIDD*ENCD);
  cvt(W_out, Wout_bf, VOCAB*HIDD);
  cvt(W_ph,  Wph_bf,  PHON*HIDD);

  init_kernel<<<32, 256, 0, stream>>>(h_bf, c_g, h2all + (size_t)MROWS*HIDD,
                                      bar);

  // enc_proj: 256 M-tiles x 32 N-tiles = 8192 waves -> 1024 blocks
  encproj_kernel<<<1024, 256, 0, stream>>>(enc_bf, Wenc_bf, b_enc, eproj);

  // persistent cooperative recurrence: 16 blocks x 256 threads
  decoder_kernel<<<Bsz, 256, 0, stream>>>(
      enc, mask, targets, emb_table, b_ih, b_hh, b_q, v_vec,
      Wq_bf, Wih_bf, Whh_bf, eproj, q_g, x_bf, h_bf, c_g, gates, h2all, bar);

  // logits: 1000 N-pairs x 4 M-groups = 4000 blocks of 8 waves (2x2 tiling)
  logits_kernel<<<(VOCAB / 32) * 4, 256, 0, stream>>>(h2all, Wout_bf, b_out,
                                                      out_logits);

  // phon: 63*3 = 189 waves -> 24 blocks
  phon_kernel<<<24, 256, 0, stream>>>(h2all, Wph_bf, b_ph, out_phon);
}
